// NonFieldLstmEncoder_26259430047973
// MI455X (gfx1250) — compile-verified, matrix operand order checked by reading, not run on hardware
//
#include <hip/hip_runtime.h>

// ---------------------------------------------------------------------------
// Problem constants (from reference): B=64, T=2048, E=256, H=256
// ---------------------------------------------------------------------------
constexpr int Bq = 64;
constexpr int Tq = 2048;
constexpr int Eq = 256;
constexpr int Hq = 256;
constexpr int NZ = 4 * Hq;   // 1024 gate columns (i | j | f | o)
constexpr int K2 = Eq + Hq;  // 512  rows of W ([x; h])

typedef __bf16 v8bf  __attribute__((ext_vector_type(8)));
typedef __bf16 v16bf __attribute__((ext_vector_type(16)));
typedef float  v8f   __attribute__((ext_vector_type(8)));

#define DEVFN __device__ __forceinline__

DEVFN v16bf cat16(v8bf lo, v8bf hi) {
  return __builtin_shufflevector(lo, hi, 0, 1, 2, 3, 4, 5, 6, 7,
                                          8, 9, 10, 11, 12, 13, 14, 15);
}

DEVFN float sigf(float x) { return 1.0f / (1.0f + __expf(-x)); }
// overflow-safe tanh: 1 - 2/(e^{2x}+1)
DEVFN float tanhf_(float x) {
  float e = __expf(2.0f * x);
  return 1.0f - 2.0f / (e + 1.0f);
}

DEVFN void store_final(const v8f& h_reg, const v8f& c_reg, int rowBase,
                       int hcol, float* __restrict__ out) {
  const size_t o0 = (size_t)Bq * Tq * Hq;
#pragma unroll
  for (int v = 0; v < 8; ++v) {
    size_t row = (size_t)(rowBase + v);
    out[o0 + row * Hq + hcol]                   = h_reg[v];
    out[o0 + (size_t)Bq * Hq + row * Hq + hcol] = c_reg[v];
  }
}

// ---------------------------------------------------------------------------
// Kernel 3a (defined FIRST so the disasm snippet shows its inner loop):
//   recurrent kernel, precomputed-x path. grid = 4 (16 batch rows each),
//   block = 512 (16 waves; wave w owns H-columns [16w,16w+16) of all gates).
//
//   WGP register file = ~4096 wave32-VGPRs = 512KB = exactly |W_h| in bf16,
//   so W_h is split to fit on-WGP with ZERO loop weight memory traffic:
//     * gate i (8 kt) + gate j (kt 0..5): registers, 14 x v16bf = 112 VGPRs
//     * gates f,o (all kt):               LDS Wlds  (264KB)
//     * gate j (kt 6..7):                 LDS Wlds2 ( 36KB)
//   Both LDS arrays are filled ONCE via global_load_async_to_lds_b128 and
//   drained with a single s_wait_asynccnt.
//   Gx (precomputed x-part incl. bias) is double-buffered in REGISTERS:
//   cur[4] for step t+1 loads while step t's WMMAs run (hides L2 latency).
// ---------------------------------------------------------------------------
__global__ void __launch_bounds__(512)
lstm_rec_precomp_kernel(const int* __restrict__ lens,   // [B]
                        const __bf16* __restrict__ Wt,  // [NZ][K2] bf16
                        const float* __restrict__ Gx,   // [B*T][NZ] fp32
                        float* __restrict__ out) {      // outputs | h | c
  constexpr int PAD   = 8;
  constexpr int WROW  = (Hq + PAD) * 2;  // 528B row stride (bank spread)
  constexpr int W2ROW = (64 + PAD) * 2;  // 144B row stride
  __shared__ __attribute__((aligned(16))) __bf16 As[16][Hq + PAD];     // 8.3KB
  __shared__ __attribute__((aligned(16))) __bf16 Wlds[512][Hq + PAD];  // 264KB
  __shared__ __attribute__((aligned(16))) __bf16 Wlds2[256][64 + PAD]; //  36KB

  const int tid  = threadIdx.x;
  const int lane = tid & 31;
  const int wv   = tid >> 5;       // 0..15 : H tile owned by this wave
  const int nl   = lane & 15;
  const int hi   = lane >> 4;
  const int b0   = blockIdx.x * 16;
  const int hcol = wv * 16 + nl;   // column within H (0..255)
  const int rowBase = b0 + 8 * hi;

  // zero h tile (h_0 = 0)
  for (int i = tid; i < 16 * Hq; i += 512) {
    int r = i >> 8, c = i & 255;
    As[r][c] = (__bf16)0.0f;
  }

  const unsigned long long wtBase = (unsigned long long)(uintptr_t)Wt;

  // ---- async fill of Wlds: gates {f,o} x K(h) of Wt (once) ----
  // Wlds row rr <-> z-column 512+rr, K columns 256..511 of Wt.
  {
    const int rr = tid;  // 0..511
#pragma unroll
    for (int c = 0; c < 32; ++c) {   // 32 x 16B = 512B per row
      unsigned lds  = (unsigned)(uintptr_t)&Wlds[rr][c * 8];
      unsigned goff = (unsigned)((512 + rr) * (K2 * 2) + Eq * 2 + c * 16);
      asm volatile("global_load_async_to_lds_b128 %0, %1, %2"
                   :: "v"(lds), "v"(goff), "s"(wtBase) : "memory");
    }
  }
  // ---- async fill of Wlds2: gate j, K(h) tiles 6..7 (K 448..511 of Wt) ----
  // Wlds2 row rr2 <-> z-column 256+rr2, 128B per row.
  {
    const int rr2  = tid >> 1;      // 0..255
    const int half = tid & 1;       // 64B halves
#pragma unroll
    for (int c = 0; c < 4; ++c) {   // 4 x 16B = 64B per half-row
      unsigned lds  = (unsigned)(uintptr_t)&Wlds2[rr2][half * 32 + c * 8];
      unsigned goff = (unsigned)((256 + rr2) * (K2 * 2) + (Eq + 192) * 2 +
                                 half * 64 + c * 16);
      asm volatile("global_load_async_to_lds_b128 %0, %1, %2"
                   :: "v"(lds), "v"(goff), "s"(wtBase) : "memory");
    }
  }

  int len_r[8];
#pragma unroll
  for (int v = 0; v < 8; ++v) len_r[v] = lens[rowBase + v];

  // ---- register-resident weights: gate i (8 kt) + gate j (kt 0..5) ----
  v16bf wregi[8];
#pragma unroll
  for (int kt = 0; kt < 8; ++kt) {
    const char* Bb = (const char*)Wt + (size_t)hcol * (K2 * 2) +
                     (Eq + kt * 32) * 2;
    wregi[kt] = cat16(*(const v8bf*)(Bb + hi * 16),
                      *(const v8bf*)(Bb + 32 + hi * 16));
  }
  v16bf wregj[6];
#pragma unroll
  for (int kt = 0; kt < 6; ++kt) {
    const char* Bb = (const char*)Wt + (size_t)(256 + hcol) * (K2 * 2) +
                     (Eq + kt * 32) * 2;
    wregj[kt] = cat16(*(const v8bf*)(Bb + hi * 16),
                      *(const v8bf*)(Bb + 32 + hi * 16));
  }

  // ---- register double buffer of Gx: preload step 0 ----
  v8f cur[4];
#pragma unroll
  for (int g = 0; g < 4; ++g) {
    const float* gp = Gx + ((size_t)rowBase * Tq + 0) * NZ + g * 256 + hcol;
#pragma unroll
    for (int v = 0; v < 8; ++v) cur[g][v] = gp[(size_t)v * Tq * NZ];
  }

  v8f h_reg = {}, c_reg = {};
  const char* Ab  = (const char*)&As[0][0];
  const char* Wb  = (const char*)&Wlds[0][0];
  const char* W2b = (const char*)&Wlds2[0][0];
  const int   Arow = (Hq + PAD) * 2;

  asm volatile("s_wait_asynccnt 0x0" ::: "memory");  // LDS weight fills done

  for (int t = 0; t < Tq; ++t) {
    __syncthreads();  // weights (t=0) / h tile from last step visible

    v8f acc[4];
#pragma unroll
    for (int g = 0; g < 4; ++g) acc[g] = cur[g];

    // start Gx loads for step t+1 (a full iteration to complete)
    if (t + 1 < Tq) {
#pragma unroll
      for (int g = 0; g < 4; ++g) {
        const float* gp =
            Gx + ((size_t)rowBase * Tq + (t + 1)) * NZ + g * 256 + hcol;
#pragma unroll
        for (int v = 0; v < 8; ++v) cur[g][v] = gp[(size_t)v * Tq * NZ];
      }
    }

    // ---- WMMA K-loop over h (K = 256) ----
#pragma unroll
    for (int kt = 0; kt < 8; ++kt) {
      v8bf a0 = *(const v8bf*)(Ab + nl * Arow + kt * 64 + hi * 16);
      v8bf a1 = *(const v8bf*)(Ab + nl * Arow + kt * 64 + 32 + hi * 16);
      v16bf av = cat16(a0, a1);
      // gate i : registers
      acc[0] = __builtin_amdgcn_wmma_f32_16x16x32_bf16(
          false, av, false, wregi[kt], (short)0, acc[0], false, false);
      // gate j : registers (kt<6) or Wlds2 (kt 6,7) — folded at compile time
      v16bf bj;
      if (kt < 6) {
        bj = wregj[kt];
      } else {
        const char* Bb2 = W2b + (size_t)hcol * (size_t)W2ROW + (kt - 6) * 64;
        bj = cat16(*(const v8bf*)(Bb2 + hi * 16),
                   *(const v8bf*)(Bb2 + 32 + hi * 16));
      }
      acc[1] = __builtin_amdgcn_wmma_f32_16x16x32_bf16(
          false, av, false, bj, (short)0, acc[1], false, false);
      // gates f,o : Wlds
#pragma unroll
      for (int g2 = 0; g2 < 2; ++g2) {
        const char* Bb = Wb + (size_t)(g2 * 256 + hcol) * (size_t)WROW + kt * 64;
        v8bf w0 = *(const v8bf*)(Bb + hi * 16);
        v8bf w1 = *(const v8bf*)(Bb + 32 + hi * 16);
        acc[2 + g2] = __builtin_amdgcn_wmma_f32_16x16x32_bf16(
            false, av, false, cat16(w0, w1), (short)0, acc[2 + g2], false, false);
      }
    }

    // ---- gates, masked update, output store ----
#pragma unroll
    for (int v = 0; v < 8; ++v) {
      float iv = acc[0][v], jv = acc[1][v], fv = acc[2][v], ov = acc[3][v];
      float cn = c_reg[v] * sigf(fv + 1.0f) + sigf(iv) * tanhf_(jv);
      float hn = sigf(ov) * tanhf_(cn);
      bool act = t < len_r[v];
      h_reg[v] = act ? hn : h_reg[v];
      c_reg[v] = act ? cn : c_reg[v];
      out[((size_t)(rowBase + v) * Tq + t) * Hq + hcol] = act ? hn : 0.0f;
    }

    __syncthreads();  // everyone done reading old h before overwrite
#pragma unroll
    for (int v = 0; v < 8; ++v)
      As[8 * hi + v][hcol] = (__bf16)h_reg[v];
  }

  store_final(h_reg, c_reg, rowBase, hcol, out);
}

// ---------------------------------------------------------------------------
// Kernel 1: transpose+convert W (512x1024 f32 row-major)
//           -> Wt (1024x512 bf16 row-major, K contiguous per z-column)
// ---------------------------------------------------------------------------
__global__ void __launch_bounds__(256)
wt_convert_kernel(const float* __restrict__ W, __bf16* __restrict__ Wt) {
  int idx = blockIdx.x * 256 + threadIdx.x;  // 0 .. 512*1024-1
  int n = idx >> 9;                          // z column 0..1023
  int k = idx & 511;                         // K row    0..511
  Wt[(size_t)n * K2 + k] = (__bf16)W[(size_t)k * NZ + n];
}

// ---------------------------------------------------------------------------
// Kernel 2: full-chip GEMM  Gx[b*T+t][:] = x[b,t,:] @ W_x + bias
//   grid = (B*T/16, NZ/128), block = 256 (8 waves, one 16x16 N-tile per wave)
// ---------------------------------------------------------------------------
__global__ void __launch_bounds__(256)
gemm_x_kernel(const float* __restrict__ X,    // [B*T][E] fp32
              const __bf16* __restrict__ Wt,  // [NZ][K2] bf16
              const float* __restrict__ bias, // [NZ]
              float* __restrict__ Gx) {       // [B*T][NZ] fp32
  constexpr int PAD = 8;                      // 16B row pad -> bank spread
  __shared__ __attribute__((aligned(16))) __bf16 As[16][Eq + PAD];

  const int tid  = threadIdx.x;
  const int lane = tid & 31;
  const int wv   = tid >> 5;
  const size_t mrow0 = (size_t)blockIdx.x * 16;

  {  // cooperative load+convert of the 16x256 fp32 A tile -> bf16 LDS
    int r  = tid >> 4;
    int c0 = (tid & 15) * 16;
    const float4* s4 = (const float4*)(X + (mrow0 + r) * (size_t)Eq + c0);
    float4 f0 = s4[0], f1 = s4[1], f2 = s4[2], f3 = s4[3];
    v8bf o0, o1;
    o0[0] = (__bf16)f0.x; o0[1] = (__bf16)f0.y; o0[2] = (__bf16)f0.z; o0[3] = (__bf16)f0.w;
    o0[4] = (__bf16)f1.x; o0[5] = (__bf16)f1.y; o0[6] = (__bf16)f1.z; o0[7] = (__bf16)f1.w;
    o1[0] = (__bf16)f2.x; o1[1] = (__bf16)f2.y; o1[2] = (__bf16)f2.z; o1[3] = (__bf16)f2.w;
    o1[4] = (__bf16)f3.x; o1[5] = (__bf16)f3.y; o1[6] = (__bf16)f3.z; o1[7] = (__bf16)f3.w;
    *(v8bf*)&As[r][c0]     = o0;
    *(v8bf*)&As[r][c0 + 8] = o1;
  }
  __syncthreads();

  const int nl = lane & 15, hi = lane >> 4;
  const int zcol = blockIdx.y * 128 + wv * 16 + nl;
  const float bb = bias[zcol];
  v8f acc = {bb, bb, bb, bb, bb, bb, bb, bb};

  const char* Ab = (const char*)&As[0][0];
  const int   Arow = (Eq + PAD) * 2;                       // A row stride bytes
  const char* Bb = (const char*)Wt + (size_t)zcol * (K2 * 2);

#pragma unroll
  for (int kt = 0; kt < Eq / 32; ++kt) {                   // K = 256
    v8bf a0 = *(const v8bf*)(Ab + nl * Arow + kt * 64 + hi * 16);
    v8bf a1 = *(const v8bf*)(Ab + nl * Arow + kt * 64 + 32 + hi * 16);
    v8bf b0 = *(const v8bf*)(Bb + kt * 64 + hi * 16);
    v8bf b1 = *(const v8bf*)(Bb + kt * 64 + 32 + hi * 16);
    acc = __builtin_amdgcn_wmma_f32_16x16x32_bf16(
        false, cat16(a0, a1), false, cat16(b0, b1), (short)0, acc, false, false);
  }

  float* dst = Gx + (mrow0 + (size_t)hi * 8) * NZ + zcol;
#pragma unroll
  for (int v = 0; v < 8; ++v) dst[(size_t)v * NZ] = acc[v];
}

// ---------------------------------------------------------------------------
// Kernel 3b: fused fallback (small workspace): x converted into LDS each
//   step, WMMA K-loop over [x; h] (K = 512), W streamed from L2.
// ---------------------------------------------------------------------------
__global__ void __launch_bounds__(512)
lstm_rec_fused_kernel(const float* __restrict__ X,    // [B][T][E] fp32
                      const int* __restrict__ lens,   // [B]
                      const __bf16* __restrict__ Wt,  // [NZ][K2] bf16
                      const float* __restrict__ bias, // [NZ]
                      float* __restrict__ out) {
  constexpr int PAD = 8;
  __shared__ __attribute__((aligned(16))) __bf16 As[16][K2 + PAD];

  const int tid  = threadIdx.x;
  const int lane = tid & 31;
  const int wv   = tid >> 5;
  const int nl   = lane & 15;
  const int hi   = lane >> 4;
  const int b0   = blockIdx.x * 16;
  const int hcol = wv * 16 + nl;
  const int rowBase = b0 + 8 * hi;

  for (int i = tid; i < 16 * Hq; i += 512) {
    int r = i >> 8, c = i & 255;
    As[r][Eq + c] = (__bf16)0.0f;
  }

  int len_r[8];
#pragma unroll
  for (int v = 0; v < 8; ++v) len_r[v] = lens[rowBase + v];

  v8f h_reg = {}, c_reg = {};
  const char* Ab   = (const char*)&As[0][0];
  const int   Arow = (K2 + PAD) * 2;

  __syncthreads();

  for (int t = 0; t < Tq; ++t) {
    {  // cooperative fp32->bf16 load of x_t into As[:, 0:256]
      int r  = tid >> 5;
      int c0 = (tid & 31) * 8;
      const float4* s4 =
          (const float4*)(X + ((size_t)(b0 + r) * Tq + t) * Eq + c0);
      float4 f0 = s4[0], f1 = s4[1];
      if (t + 1 < Tq)
        __builtin_prefetch((const char*)(s4) + (size_t)Eq * 4, 0, 1);
      v8bf o;
      o[0] = (__bf16)f0.x; o[1] = (__bf16)f0.y; o[2] = (__bf16)f0.z; o[3] = (__bf16)f0.w;
      o[4] = (__bf16)f1.x; o[5] = (__bf16)f1.y; o[6] = (__bf16)f1.z; o[7] = (__bf16)f1.w;
      *(v8bf*)&As[r][c0] = o;
    }
    __syncthreads();

    v8f acc[4];
#pragma unroll
    for (int g = 0; g < 4; ++g) {
      float bb = bias[g * 256 + hcol];
      acc[g] = (v8f){bb, bb, bb, bb, bb, bb, bb, bb};
    }

#pragma unroll 4
    for (int kt = 0; kt < K2 / 32; ++kt) {
      v8bf a0 = *(const v8bf*)(Ab + nl * Arow + kt * 64 + hi * 16);
      v8bf a1 = *(const v8bf*)(Ab + nl * Arow + kt * 64 + 32 + hi * 16);
      v16bf av = cat16(a0, a1);
#pragma unroll
      for (int g = 0; g < 4; ++g) {
        const char* Bb = (const char*)Wt +
                         (size_t)(g * 256 + hcol) * (K2 * 2) + kt * 64;
        v8bf w0 = *(const v8bf*)(Bb + hi * 16);
        v8bf w1 = *(const v8bf*)(Bb + 32 + hi * 16);
        acc[g] = __builtin_amdgcn_wmma_f32_16x16x32_bf16(
            false, av, false, cat16(w0, w1), (short)0, acc[g], false, false);
      }
    }

#pragma unroll
    for (int v = 0; v < 8; ++v) {
      float iv = acc[0][v], jv = acc[1][v], fv = acc[2][v], ov = acc[3][v];
      float cn = c_reg[v] * sigf(fv + 1.0f) + sigf(iv) * tanhf_(jv);
      float hn = sigf(ov) * tanhf_(cn);
      bool act = t < len_r[v];
      h_reg[v] = act ? hn : h_reg[v];
      c_reg[v] = act ? cn : c_reg[v];
      out[((size_t)(rowBase + v) * Tq + t) * Hq + hcol] = act ? hn : 0.0f;
    }

    __syncthreads();
#pragma unroll
    for (int v = 0; v < 8; ++v)
      As[8 * hi + v][Eq + hcol] = (__bf16)h_reg[v];
  }

  store_final(h_reg, c_reg, rowBase, hcol, out);
}

// ---------------------------------------------------------------------------
// Host launcher
// ---------------------------------------------------------------------------
extern "C" void kernel_launch(void* const* d_in, const int* in_sizes, int n_in,
                              void* d_out, int out_size, void* d_ws,
                              size_t ws_size, hipStream_t stream) {
  const float* X    = (const float*)d_in[0];  // (B,T,E) fp32
  const int*   lens = (const int*)d_in[1];    // (B,)    i32
  const float* W    = (const float*)d_in[2];  // (E+H, 4H) fp32
  const float* bias = (const float*)d_in[3];  // (4H,)  fp32
  float*       out  = (float*)d_out;

  __bf16* Wt = (__bf16*)d_ws;
  const size_t wtBytes = (size_t)NZ * K2 * sizeof(__bf16);      // 1 MB
  const size_t gxBytes = (size_t)Bq * Tq * NZ * sizeof(float);  // 512 MB

  wt_convert_kernel<<<(NZ * K2) / 256, 256, 0, stream>>>(W, Wt);

  if (ws_size >= wtBytes + gxBytes) {
    float* Gx = (float*)((char*)d_ws + wtBytes);
    dim3 grid((Bq * Tq) / 16, NZ / 128);
    gemm_x_kernel<<<grid, 256, 0, stream>>>(X, Wt, bias, Gx);
    lstm_rec_precomp_kernel<<<Bq / 16, 512, 0, stream>>>(lens, Wt, Gx, out);
  } else {
    lstm_rec_fused_kernel<<<Bq / 16, 512, 0, stream>>>(X, lens, Wt, bias, out);
  }
}
